// TransformerWithRelativePosition_69879117906567
// MI455X (gfx1250) — matrix operand order repaired and checked by reading, not compiled, not run on hardware
//
#include <hip/hip_runtime.h>
#include <hip/hip_bf16.h>
#include <math.h>

// Problem constants (match reference)
#define BB   4
#define LL   1024
#define DD   512
#define HH   8
#define HDIM 64
#define II   2048
#define RELN 2049      // 2*MAXLEN+1
#define RVP  2112      // padded width of rel_v^T (covers r0&~31 + 34*32, mult of 32)

typedef __attribute__((ext_vector_type(16))) _Float16 v16h;
typedef __attribute__((ext_vector_type(8)))  _Float16 v8h;
typedef __attribute__((ext_vector_type(8)))  float    v8f;

// ---------------------------------------------------------------------------
// WMMA core: D = A(16x32 f16) * B(32x16 f16) + C(16x16 f32)
// ---------------------------------------------------------------------------
__device__ __forceinline__ v8f wmma16x16x32(v16h a, v16h b, v8f c) {
  return __builtin_amdgcn_wmma_f32_16x16x32_f16(
      /*neg_a=*/false, a, /*neg_b=*/false, b,
      /*c_mod=*/(short)0, c, /*reuse_a=*/false, /*reuse_b=*/false);
}

// Load a 16-half fragment as two aligned 8-half (16B) runs at offsets o0, o1.
// A-frag (16-bit A layout): runs at kb and 16+kb (kb = 8*(lane>>4))
// B-frag (16-bit B layout): runs at kb and 16+kb, same k order as A
__device__ __forceinline__ v16h ldfrag(const _Float16* __restrict__ p, int o0, int o1) {
  v8h lo = *(const v8h*)(p + o0);
  v8h hi = *(const v8h*)(p + o1);
  v16h r;
#pragma unroll
  for (int e = 0; e < 8; ++e) { r[e] = lo[e]; r[8 + e] = hi[e]; }
  return r;
}

// ---------------------------------------------------------------------------
// GEMM: acc = A[MxK](f16) @ Wt[NxK](f16, pre-transposed). One wave / 16x16 tile.
// Epilogues: 0=f16 out, 1=dual f16 out (+qb1/+qb2), 2=f16 transposed out (vT),
//            3=f32 out + residual, 4=silu f16 out
// ---------------------------------------------------------------------------
#define EPI_F16    0
#define EPI_QOUT   1
#define EPI_VT     2
#define EPI_F32RES 3
#define EPI_SILU16 4

template <int EPI>
__global__ __launch_bounds__(128) void gemm16(
    const _Float16* __restrict__ A, const _Float16* __restrict__ Wt,
    const float* __restrict__ bias,
    const float* __restrict__ aux0, const float* __restrict__ aux1,
    float* __restrict__ outF, _Float16* __restrict__ outH,
    _Float16* __restrict__ outH2, int M, int N, int K) {
  const int lane  = threadIdx.x & 31;
  const int m     = lane & 15;
  const int half  = lane >> 4;
  const int kb    = half * 8;
  const int kbase = half * 16;
  const int wid    = blockIdx.x * (blockDim.x >> 5) + (threadIdx.x >> 5);
  const int ntiles = N >> 4;
  if (wid >= (M >> 4) * ntiles) return;
  const int mt = wid / ntiles;
  const int nt = wid - mt * ntiles;
  const int arow = mt * 16 + m;
  const int ncol = nt * 16 + m;
  const _Float16* ap = A + (size_t)arow * K;
  const _Float16* wp = Wt + (size_t)ncol * K;
  v8f acc = {};
  for (int k0 = 0; k0 < K; k0 += 32) {
    if (k0 + 32 < K) {
      __builtin_prefetch(ap + k0 + 32, 0, 1);  // global_prefetch_b8
      __builtin_prefetch(wp + k0 + 32, 0, 1);
    }
    v16h a = ldfrag(ap + k0, kb, 16 + kb);
    v16h b = ldfrag(wp + k0, kb, 16 + kb);
    acc = wmma16x16x32(a, b, acc);
  }
  const float bv = bias[ncol];
#pragma unroll
  for (int v = 0; v < 8; ++v) {
    const int r = mt * 16 + v + 8 * half;  // C layout: M = v + 8*half, N = lane&15
    const float val = acc[v] + bv;
    if constexpr (EPI == EPI_F16) {
      outH[(size_t)r * N + ncol] = (_Float16)val;
    } else if constexpr (EPI == EPI_QOUT) {
      outH [(size_t)r * N + ncol] = (_Float16)(val + aux0[ncol]);  // q + qb1
      outH2[(size_t)r * N + ncol] = (_Float16)(val + aux1[ncol]);  // q + qb2
    } else if constexpr (EPI == EPI_VT) {
      const int b_ = r >> 10, l_ = r & (LL - 1);  // vT[b][d][l]
      outH[((size_t)b_ * DD + ncol) * LL + l_] = (_Float16)val;
    } else if constexpr (EPI == EPI_F32RES) {
      outF[(size_t)r * N + ncol] = val + aux0[(size_t)r * N + ncol];
    } else {  // EPI_SILU16
      outH[(size_t)r * N + ncol] = (_Float16)(val / (1.0f + __expf(-val)));
    }
  }
}

// ---------------------------------------------------------------------------
// Attention scores: att[b,h,i,j] = (q1.k + q2.rel_k[j-i+1024]) / 8, masked.
// One wave / 16x16 (i,j) tile; att2 via 16x32 P-tile in LDS + diagonal gather.
// ---------------------------------------------------------------------------
__global__ __launch_bounds__(128) void att_kernel(
    const _Float16* __restrict__ q1, const _Float16* __restrict__ q2,
    const _Float16* __restrict__ k, const _Float16* __restrict__ relk,
    const unsigned char* __restrict__ amask, float* __restrict__ att) {
  __shared__ float pbuf[4][16 * 32];
  const int lane  = threadIdx.x & 31;
  const int m     = lane & 15;
  const int half  = lane >> 4;
  const int kb    = half * 8;
  const int kbase = half * 16;
  const int ws    = threadIdx.x >> 5;
  const int wid   = blockIdx.x * 4 + ws;

  const int JT = LL / 16, per_bh = JT * JT;
  const int bh  = wid / per_bh;
  const int rem = wid - bh * per_bh;
  const int it = rem / JT, jt = rem - it * JT;
  const int b = bh / HH, h = bh - b * HH;
  const int i0 = it * 16, j0 = jt * 16;

  const _Float16* q1row = q1 + ((size_t)(b * LL + i0 + m)) * DD + h * HDIM;
  const _Float16* q2row = q2 + ((size_t)(b * LL + i0 + m)) * DD + h * HDIM;
  const _Float16* krow  = k  + ((size_t)(b * LL + j0 + m)) * DD + h * HDIM;
  const int base = j0 - i0 + 1009;  // r = base + c, c in [0,32); always in [1,2048]
  const _Float16* rk0 = relk + (size_t)(base + m) * HDIM;
  const _Float16* rk1 = relk + (size_t)(base + 16 + m) * HDIM;

  v8f a1 = {}, a2a = {}, a2b = {};
#pragma unroll
  for (int k0 = 0; k0 < HDIM; k0 += 32) {
    v16h aq1 = ldfrag(q1row + k0, kb, 16 + kb);
    v16h bk  = ldfrag(krow + k0, kb, 16 + kb);
    a1 = wmma16x16x32(aq1, bk, a1);
    v16h aq2 = ldfrag(q2row + k0, kb, 16 + kb);
    v16h br0 = ldfrag(rk0 + k0, kb, 16 + kb);
    a2a = wmma16x16x32(aq2, br0, a2a);
    v16h br1 = ldfrag(rk1 + k0, kb, 16 + kb);
    a2b = wmma16x16x32(aq2, br1, a2b);
  }
  // spill 16x32 P tile, then diagonal gather c = jj - ii + 15
#pragma unroll
  for (int v = 0; v < 8; ++v) {
    const int rM = v + 8 * half;
    pbuf[ws][rM * 32 + m]      = a2a[v];
    pbuf[ws][rM * 32 + 16 + m] = a2b[v];
  }
  const bool valid = amask[b * LL + j0 + m] != 0;
#pragma unroll
  for (int v = 0; v < 8; ++v) {
    const int rM = v + 8 * half;
    const float p2 = pbuf[ws][rM * 32 + (m - rM + 15)];
    float val = (a1[v] + p2) * 0.125f;  // 1/sqrt(64)
    if (!valid) val = -3.402823466e38f;
    att[((size_t)(b * HH + h) * LL + i0 + rM) * LL + j0 + m] = val;
  }
}

// ---------------------------------------------------------------------------
// Context: ctx[b,i,h,d] = probs@v + band(probs)@rel_v (fused accumulators).
// probs f16 row-major; v pre-transposed f16 [b][d][l]; rel_v^T padded [d][r].
// One wave per (b,h, i-tile, 16-wide d-tile).
// ---------------------------------------------------------------------------
__global__ __launch_bounds__(128) void ctx_kernel(
    const _Float16* __restrict__ probs, const _Float16* __restrict__ vT,
    const _Float16* __restrict__ relvT, _Float16* __restrict__ ctx) {
  const int lane  = threadIdx.x & 31;
  const int m     = lane & 15;
  const int half  = lane >> 4;
  const int kb    = half * 8;
  const int kbase = half * 16;
  const int wid = blockIdx.x * 4 + (threadIdx.x >> 5);
  const int nt = wid & 3;
  const int it = (wid >> 2) & 63;
  const int bh = wid >> 8;
  const int b = bh >> 3, h = bh & 7;
  const int i0 = it * 16;

  const _Float16* prow = probs + ((size_t)bh * LL + i0 + m) * LL;
  const int dcol = h * HDIM + nt * 16 + m;  // column in (B,L,D) ctx
  const _Float16* vrow = vT + ((size_t)b * DD + dcol) * LL;
  const _Float16* rvrow = relvT + (size_t)(nt * 16 + m) * RVP;
  v8f acc = {};

  // w1o: probs(16xL) @ v(Lx16), both contiguous f16
  for (int k0 = 0; k0 < LL; k0 += 32) {
    v16h a  = ldfrag(prow + k0, kb, 16 + kb);
    v16h bf = ldfrag(vrow + k0, kb, 16 + kb);
    acc = wmma16x16x32(a, bf, acc);
  }

  // w2o: banded shifted-probs @ rel_v ; r in [1024-i0-15, 2047-i0]
  // aligned band start => aligned B loads; padded rel_vT covers overshoot.
  const int irow = i0 + m;
  const int r0 = (1009 - i0) & ~31;  // aligned down, >= 0
  for (int t = 0; t < 34; ++t) {
    const int rb  = r0 + t * 32;
    const int off = rb + irow - 1024;  // m_idx = off + K, guarded
    v16h a;
#pragma unroll
    for (int e = 0; e < 8; ++e) {
      const int mi = off + kb + e;
      a[e] = (mi >= 0 && mi < LL) ? prow[mi] : (_Float16)0.0f;
    }
#pragma unroll
    for (int e = 0; e < 8; ++e) {
      const int mi = off + 16 + kb + e;
      a[8 + e] = (mi >= 0 && mi < LL) ? prow[mi] : (_Float16)0.0f;
    }
    v16h bf = ldfrag(rvrow + rb, kb, 16 + kb);  // zero-padded beyond 2048
    acc = wmma16x16x32(a, bf, acc);
  }

#pragma unroll
  for (int v = 0; v < 8; ++v) {
    const int row = i0 + v + 8 * half;
    ctx[(size_t)(b * LL + row) * DD + dcol] = (_Float16)acc[v];
  }
}

// ---------------------------------------------------------------------------
// LayerNorm over D=512; one 256-thread block per row (8 wave32).
// ---------------------------------------------------------------------------
template <typename OT>
__global__ __launch_bounds__(256) void ln_kernel(
    const float* __restrict__ x, const float* __restrict__ g,
    const float* __restrict__ bta, OT* __restrict__ out) {
  __shared__ float sm[8];
  const size_t row = blockIdx.x;
  const float* xr = x + row * DD;
  OT* orow = out + row * DD;
  const int t = threadIdx.x;
  const float v0 = xr[t], v1 = xr[t + 256];
  float s = v0 + v1;
#pragma unroll
  for (int o = 16; o > 0; o >>= 1) s += __shfl_xor(s, o, 32);
  if ((t & 31) == 0) sm[t >> 5] = s;
  __syncthreads();
  float tot = sm[0];
#pragma unroll
  for (int i = 1; i < 8; ++i) tot += sm[i];
  const float mean = tot * (1.0f / DD);
  const float d0 = v0 - mean, d1 = v1 - mean;
  float sq = d0 * d0 + d1 * d1;
#pragma unroll
  for (int o = 16; o > 0; o >>= 1) sq += __shfl_xor(sq, o, 32);
  __syncthreads();
  if ((t & 31) == 0) sm[t >> 5] = sq;
  __syncthreads();
  float vtot = sm[0];
#pragma unroll
  for (int i = 1; i < 8; ++i) vtot += sm[i];
  const float rstd = rsqrtf(vtot * (1.0f / DD) + 1e-5f);
  orow[t]       = (OT)(d0 * rstd * g[t] + bta[t]);
  orow[t + 256] = (OT)(d1 * rstd * g[t + 256] + bta[t + 256]);
}

// ---------------------------------------------------------------------------
// Row softmax over L=1024: reads f32 scores, writes f16 probs.
// ---------------------------------------------------------------------------
__global__ __launch_bounds__(256) void softmax_kernel(
    const float* __restrict__ att, _Float16* __restrict__ ph) {
  __shared__ float sm[8];
  const size_t row = blockIdx.x;
  const float* p = att + row * LL;
  _Float16* po = ph + row * LL;
  const int t = threadIdx.x;
  float v[4];
  float mx = -3.402823466e38f;
#pragma unroll
  for (int i = 0; i < 4; ++i) { v[i] = p[t + i * 256]; mx = fmaxf(mx, v[i]); }
#pragma unroll
  for (int o = 16; o > 0; o >>= 1) mx = fmaxf(mx, __shfl_xor(mx, o, 32));
  if ((t & 31) == 0) sm[t >> 5] = mx;
  __syncthreads();
  float bm = sm[0];
#pragma unroll
  for (int i = 1; i < 8; ++i) bm = fmaxf(bm, sm[i]);
  float s = 0.f;
#pragma unroll
  for (int i = 0; i < 4; ++i) { v[i] = __expf(v[i] - bm); s += v[i]; }
#pragma unroll
  for (int o = 16; o > 0; o >>= 1) s += __shfl_xor(s, o, 32);
  __syncthreads();
  if ((t & 31) == 0) sm[t >> 5] = s;
  __syncthreads();
  float tot = sm[0];
#pragma unroll
  for (int i = 1; i < 8; ++i) tot += sm[i];
  const float inv = 1.0f / tot;
#pragma unroll
  for (int i = 0; i < 4; ++i) po[t + i * 256] = (_Float16)(v[i] * inv);
}

// ---------------------------------------------------------------------------
// One-time conversions
// ---------------------------------------------------------------------------
// f32 (K,N) row-major -> f16 (N,K) row-major (weight transpose)
__global__ __launch_bounds__(256) void transcvt_kernel(
    const float* __restrict__ in, _Float16* __restrict__ out, int K, int N) {
  const int idx = blockIdx.x * 256 + threadIdx.x;
  if (idx >= K * N) return;
  const int kk = idx / N, n = idx - kk * N;
  out[(size_t)n * K + kk] = (_Float16)in[idx];
}

// plain f32 -> f16 copy
__global__ __launch_bounds__(256) void cvt_kernel(
    const float* __restrict__ in, _Float16* __restrict__ out, int n) {
  const int idx = blockIdx.x * 256 + threadIdx.x;
  if (idx < n) out[idx] = (_Float16)in[idx];
}

// rel_v (2049,64) f32 -> rel_v^T (64, RVP) f16, zero padded beyond r=2048
__global__ __launch_bounds__(256) void relvt_kernel(
    const float* __restrict__ relv, _Float16* __restrict__ out) {
  const int idx = blockIdx.x * 256 + threadIdx.x;
  if (idx >= HDIM * RVP) return;
  const int d = idx / RVP, r = idx - d * RVP;
  out[idx] = (r < RELN) ? (_Float16)relv[(size_t)r * HDIM + d] : (_Float16)0.0f;
}

// ---------------------------------------------------------------------------
extern "C" void kernel_launch(void* const* d_in, const int* in_sizes, int n_in,
                              void* d_out, int out_size, void* d_ws, size_t ws_size,
                              hipStream_t stream) {
  const float* x      = (const float*)d_in[0];
  const unsigned char* amask = (const unsigned char*)d_in[1];
  const float* ln1_g  = (const float*)d_in[2];
  const float* ln1_b  = (const float*)d_in[3];
  const float* wq     = (const float*)d_in[4];
  const float* bq     = (const float*)d_in[5];
  const float* wk     = (const float*)d_in[6];
  const float* bk     = (const float*)d_in[7];
  const float* wv     = (const float*)d_in[8];
  const float* bv     = (const float*)d_in[9];
  const float* qb1    = (const float*)d_in[10];
  const float* qb2    = (const float*)d_in[11];
  const float* rel_k  = (const float*)d_in[12];
  const float* rel_v  = (const float*)d_in[13];
  const float* wo     = (const float*)d_in[14];
  const float* bo     = (const float*)d_in[15];
  const float* ln2_g  = (const float*)d_in[16];
  const float* ln2_b  = (const float*)d_in[17];
  const float* fln_g  = (const float*)d_in[18];
  const float* fln_b  = (const float*)d_in[19];
  const float* w_in   = (const float*)d_in[20];
  const float* b_in   = (const float*)d_in[21];
  const float* w_out  = (const float*)d_in[22];
  const float* b_out  = (const float*)d_in[23];
  float* out = (float*)d_out;

  const int M = BB * LL;  // 4096 token rows
  typedef _Float16 h16;

  // ---- workspace layout (bytes, 256B aligned bump allocator) ----
  char* wp_ = (char*)d_ws;
  auto alloc = [&](size_t bytes) {
    char* r = wp_;
    wp_ += (bytes + 255) & ~(size_t)255;
    return r;
  };
  float* attF  = (float*)alloc((size_t)BB * HH * LL * LL * 4);  // 128 MB
  h16* probsH  = (h16*)alloc((size_t)BB * HH * LL * LL * 2);    // 64 MB
  h16* hf      = (h16*)alloc((size_t)M * DD * 2);
  h16* q1f     = (h16*)alloc((size_t)M * DD * 2);
  h16* q2f     = (h16*)alloc((size_t)M * DD * 2);
  h16* kf      = (h16*)alloc((size_t)M * DD * 2);
  h16* vT      = (h16*)alloc((size_t)M * DD * 2);
  h16* wqT     = (h16*)alloc((size_t)DD * DD * 2);
  h16* wkT     = (h16*)alloc((size_t)DD * DD * 2);
  h16* wvT     = (h16*)alloc((size_t)DD * DD * 2);
  h16* woT     = (h16*)alloc((size_t)DD * DD * 2);
  h16* w_inT   = (h16*)alloc((size_t)DD * II * 2);
  h16* w_outT  = (h16*)alloc((size_t)DD * II * 2);
  h16* rel_kf  = (h16*)alloc((size_t)RELN * HDIM * 2);
  h16* rel_vT  = (h16*)alloc((size_t)HDIM * RVP * 2);
  // attF region is dead after softmax: reuse it for the tail buffers
  char* reuse = (char*)attF;
  h16*   ctxH = (h16*)(reuse);                       //  4 MB
  float* x2   = (float*)(reuse + (8ull << 20));      //  8 MB
  float* h2   = (float*)(reuse + (16ull << 20));     //  8 MB
  h16*   fH   = (h16*)(reuse + (24ull << 20));       //  4 MB
  h16*   midH = (h16*)(reuse + (32ull << 20));       // 16 MB

  // ---- one-time weight / table conversions ----
  transcvt_kernel<<<(DD * DD + 255) / 256, 256, 0, stream>>>(wq, wqT, DD, DD);
  transcvt_kernel<<<(DD * DD + 255) / 256, 256, 0, stream>>>(wk, wkT, DD, DD);
  transcvt_kernel<<<(DD * DD + 255) / 256, 256, 0, stream>>>(wv, wvT, DD, DD);
  transcvt_kernel<<<(DD * DD + 255) / 256, 256, 0, stream>>>(wo, woT, DD, DD);
  transcvt_kernel<<<(DD * II + 255) / 256, 256, 0, stream>>>(w_in, w_inT, DD, II);
  transcvt_kernel<<<(DD * II + 255) / 256, 256, 0, stream>>>(w_out, w_outT, II, DD);
  cvt_kernel<<<(RELN * HDIM + 255) / 256, 256, 0, stream>>>(rel_k, rel_kf, RELN * HDIM);
  relvt_kernel<<<(HDIM * RVP + 255) / 256, 256, 0, stream>>>(rel_v, rel_vT);

  const dim3 blk(128);
  const int gq = (M / 16) * (DD / 16) / 4;  // 2048 blocks (1 wave / 16x16 tile)

  // ---- attention sub-layer ----
  ln_kernel<h16><<<M, 256, 0, stream>>>(x, ln1_g, ln1_b, hf);
  gemm16<EPI_QOUT><<<gq, blk, 0, stream>>>(hf, wqT, bq, qb1, qb2,
                                           nullptr, q1f, q2f, M, DD, DD);
  gemm16<EPI_F16><<<gq, blk, 0, stream>>>(hf, wkT, bk, nullptr, nullptr,
                                          nullptr, kf, nullptr, M, DD, DD);
  gemm16<EPI_VT><<<gq, blk, 0, stream>>>(hf, wvT, bv, nullptr, nullptr,
                                         nullptr, vT, nullptr, M, DD, DD);
  att_kernel<<<BB * HH * (LL / 16) * (LL / 16) / 4, blk, 0, stream>>>(
      q1f, q2f, kf, rel_kf, amask, attF);
  softmax_kernel<<<BB * HH * LL, 256, 0, stream>>>(attF, probsH);
  ctx_kernel<<<BB * HH * (LL / 16) * 4 / 4, blk, 0, stream>>>(
      probsH, vT, rel_vT, ctxH);
  gemm16<EPI_F32RES><<<gq, blk, 0, stream>>>(ctxH, woT, bo, x, nullptr,
                                             x2, nullptr, nullptr, M, DD, DD);

  // ---- FFN sub-layer ----
  ln_kernel<float><<<M, 256, 0, stream>>>(x2, ln2_g, ln2_b, h2);
  ln_kernel<h16><<<M, 256, 0, stream>>>(h2, fln_g, fln_b, fH);
  gemm16<EPI_SILU16><<<(M / 16) * (II / 16) / 4, blk, 0, stream>>>(
      fH, w_inT, b_in, nullptr, nullptr, nullptr, midH, nullptr, M, II, DD);
  gemm16<EPI_F32RES><<<gq, blk, 0, stream>>>(midH, w_outT, b_out, x2, nullptr,
                                             out, nullptr, nullptr, M, DD, II);
}
